// GCN_33380485825193
// MI455X (gfx1250) — matrix operand order repaired and checked by reading, not compiled, not run on hardware
//
#include <hip/hip_runtime.h>

// ---------------------------------------------------------------------------
// GCN layer for MI455X (gfx1250, wave32):
//   for s in 0..1:  tmp = bf16(X) @ bf16(W[s])   (WMMA f32_16x16x32_bf16)
//                   out[row[e]] += tmp[col[e]] * val[e]   (L2-resident atomics)
//   out = relu(out)
// tmp (51.2MB) reuses d_ws for both supports; tmp and out both fit in 192MB L2.
// ---------------------------------------------------------------------------

typedef __attribute__((ext_vector_type(16))) __bf16 v16bf;
typedef __attribute__((ext_vector_type(8)))  float  v8f;

#define D_IN  256
#define D_OUT 128
#define BM    128          // rows per block
#define BK    64           // K tile through LDS
#define SXK   (BK + 8)     // padded LDS stride (bf16 elems) to dodge bank conflicts
#define SWK   (BK + 8)

union FragBF {
    unsigned int   u32[8];
    unsigned short u16[16];
    v16bf          v;
};

__device__ inline unsigned short f2bf(float f) {
    // round-to-nearest-even f32 -> bf16
    unsigned int u = __builtin_bit_cast(unsigned int, f);
    u += 0x7FFFu + ((u >> 16) & 1u);
    return (unsigned short)(u >> 16);
}

// ---------------------------------------------------------------------------
// tmp[N,128] = X[N,256] @ W[256,128], bf16 inputs, f32 accumulate via WMMA.
// Block: 256 threads (8 waves). Wave w computes rows [16w,16w+16) of the
// 128x128 output tile, across 8 column tiles of 16.
// ---------------------------------------------------------------------------
__global__ __launch_bounds__(256)
void gcn_gemm_bf16(const float* __restrict__ x, const float* __restrict__ w,
                   float* __restrict__ tmp, int nrows) {
    __shared__ unsigned short sx[BM * SXK];      // X tile,  [r][k]   (bf16)
    __shared__ unsigned short sw[D_OUT * SWK];   // W tileT, [n][k]   (bf16)

    const int row0 = blockIdx.x * BM;
    const int tid  = threadIdx.x;
    const int wave = tid >> 5;          // 0..7
    const int lane = tid & 31;
    const int half = lane >> 4;         // 0: lanes 0-15, 1: lanes 16-31
    const int l16  = lane & 15;

    v8f acc[8];
    #pragma unroll
    for (int c = 0; c < 8; ++c) acc[c] = (v8f){0.f,0.f,0.f,0.f,0.f,0.f,0.f,0.f};

    for (int k0 = 0; k0 < D_IN; k0 += BK) {
        __syncthreads();   // protect LDS from previous iteration's readers
        // Stage X tile (128x64) as bf16.
        for (int i = tid; i < BM * BK; i += 256) {
            int r = i >> 6, k = i & (BK - 1);
            int gr = row0 + r;
            float v = (gr < nrows) ? x[(size_t)gr * D_IN + k0 + k] : 0.f;
            sx[r * SXK + k] = f2bf(v);
        }
        // Stage W tile transposed: sw[n][k] = W[k0+k][n] as bf16.
        for (int i = tid; i < BK * D_OUT; i += 256) {
            int k = i >> 7, n = i & (D_OUT - 1);
            sw[n * SWK + k] = f2bf(w[(size_t)(k0 + k) * D_OUT + n]);
        }
        __syncthreads();

        const int arow = wave * 16 + l16;   // A-fragment row within block tile
        #pragma unroll
        for (int ks = 0; ks < BK; ks += 32) {
            // A fragment (16x32 bf16, ISA 7.12.2): lanes 0-15 carry K={0..7,16..23},
            // lanes 16-31 carry K={8..15,24..31}; VGPR v holds K pair.
            FragBF a;
            const int abase = arow * SXK + ks + half * 8;
            #pragma unroll
            for (int v = 0; v < 8; ++v) {
                int koff = 2 * v + ((v >= 4) ? 8 : 0);
                a.u32[v] = *(const unsigned int*)&sx[abase + koff];
            }
            #pragma unroll
            for (int c = 0; c < 8; ++c) {
                // B fragment (32x16 bf16): lanes 0-15 carry K={0..7,16..23}, lanes 16-31
                // carry K={8..15,24..31} of column N = c*16 + l16; VGPR v holds K pair.
                FragBF b;
                const int bbase = (c * 16 + l16) * SWK + ks + half * 8;
                #pragma unroll
                for (int v = 0; v < 8; ++v)
                    b.u32[v] = *(const unsigned int*)&sw[bbase + 2 * v + ((v >= 4) ? 8 : 0)];
                acc[c] = __builtin_amdgcn_wmma_f32_16x16x32_bf16(
                    /*neg_a=*/false, a.v, /*neg_b=*/false, b.v,
                    /*c_mod=*/(short)0, acc[c],
                    /*reuse_a=*/false, /*reuse_b=*/false);
            }
        }
    }

    // C/D layout: VGPR r of lane holds element (M = r + 8*half, N = l16).
    #pragma unroll
    for (int c = 0; c < 8; ++c) {
        #pragma unroll
        for (int r = 0; r < 8; ++r) {
            int gr = row0 + wave * 16 + half * 8 + r;
            if (gr < nrows)
                tmp[(size_t)gr * D_OUT + c * 16 + l16] = acc[c][r];
        }
    }
}

// ---------------------------------------------------------------------------
// One wave per edge: lane l handles columns 4l..4l+3.
// out[row] += tmp[col] * val  (f32 atomics; tmp & out are L2-resident)
// ---------------------------------------------------------------------------
__global__ __launch_bounds__(256)
void gcn_scatter(const float* __restrict__ tmp, const int* __restrict__ erow,
                 const int* __restrict__ ecol, const float* __restrict__ eval,
                 float* __restrict__ out, int nedges) {
    const int wid  = blockIdx.x * (blockDim.x >> 5) + (threadIdx.x >> 5);
    const int lane = threadIdx.x & 31;
    if (wid >= nedges) return;

    const int   r = erow[wid];
    const int   c = ecol[wid];
    const float v = eval[wid];

    const float4 t = *(const float4*)(tmp + (size_t)c * D_OUT + lane * 4);
    float* dst = out + (size_t)r * D_OUT + lane * 4;
    atomicAdd(dst + 0, t.x * v);
    atomicAdd(dst + 1, t.y * v);
    atomicAdd(dst + 2, t.z * v);
    atomicAdd(dst + 3, t.w * v);
}

__global__ __launch_bounds__(256)
void gcn_relu(float* __restrict__ out, int n) {
    int i = blockIdx.x * blockDim.x + threadIdx.x;
    if (i < n) {
        float v = out[i];
        out[i] = v > 0.f ? v : 0.f;
    }
}

extern "C" void kernel_launch(void* const* d_in, const int* in_sizes, int n_in,
                              void* d_out, int out_size, void* d_ws, size_t ws_size,
                              hipStream_t stream) {
    const float* x     = (const float*)d_in[0];   // [N, 256]
    const float* w     = (const float*)d_in[1];   // [S, 256, 128]
    const int*   erows = (const int*)  d_in[2];   // [S, E]
    const int*   ecols = (const int*)  d_in[3];   // [S, E]
    const float* evals = (const float*)d_in[4];   // [S, E]
    float*       out   = (float*)d_out;           // [N, 128]
    float*       tmp   = (float*)d_ws;            // [N, 128] scratch, reused per support

    const int N = in_sizes[0] / D_IN;
    const int S = in_sizes[1] / (D_IN * D_OUT);
    const int E = (S > 0) ? in_sizes[2] / S : 0;

    hipMemsetAsync(d_out, 0, (size_t)out_size * sizeof(float), stream);

    const int gemm_blocks    = (N + BM - 1) / BM;
    const int scatter_blocks = (E + 7) / 8;        // 8 waves per 256-thread block

    for (int s = 0; s < S; ++s) {
        gcn_gemm_bf16<<<gemm_blocks, 256, 0, stream>>>(
            x, w + (size_t)s * D_IN * D_OUT, tmp, N);
        gcn_scatter<<<scatter_blocks, 256, 0, stream>>>(
            tmp, erows + (size_t)s * E, ecols + (size_t)s * E,
            evals + (size_t)s * E, out, E);
    }

    const int total = N * D_OUT;
    gcn_relu<<<(total + 255) / 256, 256, 0, stream>>>(out, total);
}
